// NAT_61220463837270
// MI455X (gfx1250) — compile-verified, matrix operand order checked by reading, not run on hardware
//
#include <hip/hip_runtime.h>
#include <hip/hip_bf16.h>

#define NN   4096
#define FIN  128
#define HH   8
#define FOUT 64
#define DD   512
#define NGH  128
#define G3   384   // 3*NGH

typedef __attribute__((ext_vector_type(16))) _Float16 v16h;
typedef __attribute__((ext_vector_type(8)))  _Float16 v8h;
typedef __attribute__((ext_vector_type(8)))  float    v8f;

union AV {
    v16h v;
    v8h  h[2];
    _Float16 e[16];
};

// ---------------------------------------------------------------------------
// f32 -> f16 conversion
// ---------------------------------------------------------------------------
__global__ void f32_to_f16_kernel(const float* __restrict__ in,
                                  _Float16* __restrict__ out, int n) {
    int i = blockIdx.x * blockDim.x + threadIdx.x;
    if (i < n) out[i] = (_Float16)in[i];
}

// ---------------------------------------------------------------------------
// Generic WMMA f16 GEMM:  C[M][N] = A[M][K] * B[N][K]^T (+ bias[N])
// A row-major f16, B row-major f16 (i.e. B[n][k]), C f32 row-major (ldc = N).
// One wave computes a 16(M) x 64(N) tile. grid = (N/64, M/16), block = 32.
// A layout per lane: row = lane&15, k-chunks [kb, kb+8) and [kb+16, kb+24),
// kb = 8*(lane>>4).  B layout: lane holds column (lane&15), same K striping.
// ---------------------------------------------------------------------------
__global__ void wmma_gemm_f16(const _Float16* __restrict__ A,
                              const _Float16* __restrict__ B,
                              const float* __restrict__ bias,
                              float* __restrict__ C,
                              int M, int N, int K) {
    const int cb   = blockIdx.x * 64;
    const int rb   = blockIdx.y * 16;
    const int lane = threadIdx.x;
    const int r    = lane & 15;
    const int kb   = (lane >> 4) * 8;

    v8f acc[4] = {};
    const _Float16* arow = A + (size_t)(rb + r) * K;

    for (int k0 = 0; k0 < K; k0 += 32) {
        AV a;
        a.h[0] = *(const v8h*)(arow + k0 + kb);
        a.h[1] = *(const v8h*)(arow + k0 + kb + 16);
#pragma unroll
        for (int g = 0; g < 4; ++g) {
            const _Float16* brow = B + (size_t)(cb + g * 16 + r) * K;
            AV b;
            b.h[0] = *(const v8h*)(brow + k0 + kb);
            b.h[1] = *(const v8h*)(brow + k0 + kb + 16);
            acc[g] = __builtin_amdgcn_wmma_f32_16x16x32_f16(
                false, a.v, false, b.v, (short)0, acc[g], false, false);
        }
    }

    const int hi8 = (lane >> 4) << 3;  // 0 or 8
    if (bias) {
#pragma unroll
        for (int g = 0; g < 4; ++g) {
            float bv = bias[cb + g * 16 + r];
#pragma unroll
            for (int i = 0; i < 8; ++i)
                C[(size_t)(rb + i + hi8) * N + cb + g * 16 + r] = acc[g][i] + bv;
        }
    } else {
#pragma unroll
        for (int g = 0; g < 4; ++g)
#pragma unroll
            for (int i = 0; i < 8; ++i)
                C[(size_t)(rb + i + hi8) * N + cb + g * 16 + r] = acc[g][i];
    }
}

// ---------------------------------------------------------------------------
// s_src[h][n] = dot(proj[n, h*64 : h*64+64], a_src[h]);  same for s_tgt
// ---------------------------------------------------------------------------
__global__ void compute_s_kernel(const float* __restrict__ proj,
                                 const float* __restrict__ a_src,
                                 const float* __restrict__ a_tgt,
                                 float* __restrict__ s_src,
                                 float* __restrict__ s_tgt) {
    int idx = blockIdx.x * blockDim.x + threadIdx.x;   // h*NN + n, n fastest
    int n = idx & (NN - 1);
    int h = idx >> 12;
    const float* pr = proj + (size_t)n * DD + h * FOUT;
    const float* as = a_src + h * FOUT;
    const float* at = a_tgt + h * FOUT;
    float sa = 0.f, sb = 0.f;
#pragma unroll 8
    for (int f = 0; f < FOUT; ++f) {
        float p = pr[f];
        sa += p * as[f];
        sb += p * at[f];
    }
    s_src[h * NN + n] = sa;
    s_tgt[h * NN + n] = sb;
}

// ---------------------------------------------------------------------------
// proj_f32[n][512] -> projT_f16[d][4096]   (16x16 LDS tiles)
// grid = (NN/16, DD/16), block = 256
// ---------------------------------------------------------------------------
__global__ void transpose_f16_kernel(const float* __restrict__ src,
                                     _Float16* __restrict__ dst) {
    __shared__ float tile[16][17];
    int nb = blockIdx.x * 16;
    int db = blockIdx.y * 16;
    int tx = threadIdx.x & 15;
    int ty = threadIdx.x >> 4;
    tile[ty][tx] = src[(size_t)(nb + ty) * DD + db + tx];
    __syncthreads();
    dst[(size_t)(db + ty) * NN + nb + tx] = (_Float16)tile[tx][ty];
}

// ---------------------------------------------------------------------------
// Flash-style GAT attention: one wave per (head, 16-row block).
// Online softmax over 4096 columns in K=32 steps; PV via WMMA f16.
// The connectivity mask is the only HBM-resident stream (64 MB); prefetch it
// ~8 iterations ahead (global_prefetch_b8 pulls the full cacheline).
// Finalize: /rowsum + skip + bias, ELU -> gat_out (f32 + f16 copy).
// grid = (NN/16, HH), block = 32.
// ---------------------------------------------------------------------------
__global__ void gat_attention_kernel(const _Float16* __restrict__ projT, // [DD][NN]
                                     const float* __restrict__ skip,     // [NN][DD]
                                     const float* __restrict__ mask,     // [NN][NN]
                                     const float* __restrict__ s_src,    // [HH][NN]
                                     const float* __restrict__ s_tgt,    // [HH][NN]
                                     const float* __restrict__ bias,     // [DD]
                                     float* __restrict__ gat_out,        // [NN][DD]
                                     _Float16* __restrict__ gat_out16) { // [NN][DD]
    const int h    = blockIdx.y;
    const int i0   = blockIdx.x * 16;
    const int lane = threadIdx.x;
    const int r    = lane & 15;          // my row within tile (A layout)
    const int kb   = (lane >> 4) * 8;    // my K sub-base (A layout)
    const int hi8  = (lane >> 4) << 3;

    const float ssrc = s_src[h * NN + i0 + r];
    const float* strow = s_tgt + h * NN;
    const float* mrow  = mask + (size_t)(i0 + r) * NN;

    float m_run = -INFINITY;
    float l_run = 0.f;
    v8f acc[4] = {};

    for (int j0 = 0; j0 < NN; j0 += 32) {
        // speculative stream-prefetch of the mask ~1KB/row ahead
        if (j0 + 256 < NN)
            __builtin_prefetch(mrow + j0 + 256 + kb, 0, 0);

        // load s_tgt + mask for my 16 columns: k in [kb,kb+8) U [kb+16,kb+24)
        float4 t0 = *(const float4*)(strow + j0 + kb);
        float4 t1 = *(const float4*)(strow + j0 + kb + 4);
        float4 t2 = *(const float4*)(strow + j0 + kb + 16);
        float4 t3 = *(const float4*)(strow + j0 + kb + 20);
        float4 m0 = *(const float4*)(mrow + j0 + kb);
        float4 m1 = *(const float4*)(mrow + j0 + kb + 4);
        float4 m2 = *(const float4*)(mrow + j0 + kb + 16);
        float4 m3 = *(const float4*)(mrow + j0 + kb + 20);

        float st[16] = {t0.x, t0.y, t0.z, t0.w, t1.x, t1.y, t1.z, t1.w,
                        t2.x, t2.y, t2.z, t2.w, t3.x, t3.y, t3.z, t3.w};
        float mk[16] = {m0.x, m0.y, m0.z, m0.w, m1.x, m1.y, m1.z, m1.w,
                        m2.x, m2.y, m2.z, m2.w, m3.x, m3.y, m3.z, m3.w};

        float sc[16];
#pragma unroll
        for (int t = 0; t < 16; ++t) {
            float s = ssrc + st[t];
            s = (s > 0.f) ? s : 0.2f * s;      // leaky_relu(0.2)
            sc[t] = s + mk[t];
        }

        // tile row-max (own 16 cols, then partner lane's 16)
        float mt = sc[0];
#pragma unroll
        for (int t = 1; t < 16; ++t) mt = fmaxf(mt, sc[t]);
        mt = fmaxf(mt, __shfl_xor(mt, 16, 32));
        float m_new = fmaxf(m_run, mt);
        float scale = __expf(m_run - m_new);   // exp(-inf)=0 on first iter

        AV a;
        float psum = 0.f;
#pragma unroll
        for (int t = 0; t < 16; ++t) {
            float p = __expf(sc[t] - m_new);
            psum += p;
            a.e[t] = (_Float16)p;              // lands directly in A layout
        }
        psum += __shfl_xor(psum, 16, 32);
        l_run = l_run * scale + psum;
        m_run = m_new;

        // rescale accumulators: C layout row of elem i = i + 8*(lane>=16)
#pragma unroll
        for (int i = 0; i < 8; ++i) {
            float si = __shfl(scale, i + hi8, 32);
#pragma unroll
            for (int g = 0; g < 4; ++g) acc[g][i] = acc[g][i] * si;
        }

        // PV: B[k][n] = projT[h*64 + g*16 + n][j0 + k]
#pragma unroll
        for (int g = 0; g < 4; ++g) {
            const _Float16* brow = projT + (size_t)(h * FOUT + g * 16 + r) * NN + j0;
            AV b;
            b.h[0] = *(const v8h*)(brow + kb);
            b.h[1] = *(const v8h*)(brow + kb + 16);
            acc[g] = __builtin_amdgcn_wmma_f32_16x16x32_f16(
                false, a.v, false, b.v, (short)0, acc[g], false, false);
        }
    }

    // finalize: divide by rowsum, + skip + bias, ELU
#pragma unroll
    for (int i = 0; i < 8; ++i) {
        float li = __shfl(l_run, i + hi8, 32);
        float inv = 1.0f / li;
        int row = i0 + i + hi8;
#pragma unroll
        for (int g = 0; g < 4; ++g) {
            int col = h * FOUT + g * 16 + r;
            float v = acc[g][i] * inv + skip[(size_t)row * DD + col] + bias[col];
            float e = (v > 0.f) ? v : (__expf(v) - 1.0f);   // elu
            gat_out[(size_t)row * DD + col]  = e;
            gat_out16[(size_t)row * DD + col] = (_Float16)e;
        }
    }
}

// ---------------------------------------------------------------------------
// GRU gate math: h_new = (1-z)*tanh(i_n + r*h_n) + z*h0
// ---------------------------------------------------------------------------
__global__ void gru_gate_kernel(const float* __restrict__ gi,   // [NN][384]
                                const float* __restrict__ gh,   // [NN][384]
                                const float* __restrict__ h0,   // [NN][128]
                                float* __restrict__ h_new) {    // [NN][128]
    int idx = blockIdx.x * blockDim.x + threadIdx.x;
    if (idx >= NN * NGH) return;
    int n = idx >> 7;
    int j = idx & (NGH - 1);
    const float* gir = gi + (size_t)n * G3;
    const float* ghr = gh + (size_t)n * G3;
    float ir = gir[j], iz = gir[NGH + j], in_ = gir[2 * NGH + j];
    float hr = ghr[j], hz = ghr[NGH + j], hn = ghr[2 * NGH + j];
    float rg = 1.f / (1.f + __expf(-(ir + hr)));
    float zg = 1.f / (1.f + __expf(-(iz + hz)));
    float ng = tanhf(in_ + rg * hn);
    h_new[idx] = (1.f - zg) * ng + zg * h0[idx];
}

// ---------------------------------------------------------------------------
// host launcher
// ---------------------------------------------------------------------------
extern "C" void kernel_launch(void* const* d_in, const int* in_sizes, int n_in,
                              void* d_out, int out_size, void* d_ws, size_t ws_size,
                              hipStream_t stream) {
    const float* x      = (const float*)d_in[0];
    const float* mask   = (const float*)d_in[1];
    const float* W_proj = (const float*)d_in[2];
    const float* a_src  = (const float*)d_in[3];
    const float* a_tgt  = (const float*)d_in[4];
    const float* bias   = (const float*)d_in[5];
    const float* W_skip = (const float*)d_in[6];
    const float* w_ih   = (const float*)d_in[7];
    const float* w_hh   = (const float*)d_in[8];
    const float* b_ih   = (const float*)d_in[9];
    const float* b_hh   = (const float*)d_in[10];
    const float* h0     = (const float*)d_in[11];

    float* gat_out = (float*)d_out;                      // [NN][DD]
    float* h_new   = gat_out + (size_t)NN * DD;          // [NN][NGH]

    char* wsp = (char*)d_ws;
    auto carve = [&](size_t bytes) -> char* {
        char* p = wsp;
        wsp += (bytes + 255) & ~(size_t)255;
        return p;
    };
    _Float16* x16     = (_Float16*)carve((size_t)NN * FIN * 2);
    _Float16* wproj16 = (_Float16*)carve((size_t)DD * FIN * 2);
    _Float16* wskip16 = (_Float16*)carve((size_t)DD * FIN * 2);
    _Float16* wih16   = (_Float16*)carve((size_t)G3 * DD * 2);
    _Float16* whh16   = (_Float16*)carve((size_t)G3 * NGH * 2);
    _Float16* h016    = (_Float16*)carve((size_t)NN * NGH * 2);
    float*    proj    = (float*)carve((size_t)NN * DD * 4);
    float*    skip    = (float*)carve((size_t)NN * DD * 4);
    _Float16* projT16 = (_Float16*)carve((size_t)DD * NN * 2);
    float*    ssrc    = (float*)carve((size_t)HH * NN * 4);
    float*    stgt    = (float*)carve((size_t)HH * NN * 4);
    _Float16* gat16   = (_Float16*)carve((size_t)NN * DD * 2);
    float*    gi      = (float*)carve((size_t)NN * G3 * 4);
    float*    gh      = (float*)carve((size_t)NN * G3 * 4);

    // 1) f16 conversions
    auto cvt = [&](const float* src, _Float16* dst, int n) {
        f32_to_f16_kernel<<<(n + 255) / 256, 256, 0, stream>>>(src, dst, n);
    };
    cvt(x, x16, NN * FIN);
    cvt(W_proj, wproj16, DD * FIN);
    cvt(W_skip, wskip16, DD * FIN);
    cvt(w_ih, wih16, G3 * DD);
    cvt(w_hh, whh16, G3 * NGH);
    cvt(h0, h016, NN * NGH);

    // 2) proj = x @ W_proj^T ; skip = x @ W_skip^T   (WMMA)
    wmma_gemm_f16<<<dim3(DD / 64, NN / 16), 32, 0, stream>>>(
        x16, wproj16, nullptr, proj, NN, DD, FIN);
    wmma_gemm_f16<<<dim3(DD / 64, NN / 16), 32, 0, stream>>>(
        x16, wskip16, nullptr, skip, NN, DD, FIN);

    // 3) attention logits s_src / s_tgt
    compute_s_kernel<<<(HH * NN) / 256, 256, 0, stream>>>(proj, a_src, a_tgt, ssrc, stgt);

    // 4) projT_f16 for WMMA-B operand
    transpose_f16_kernel<<<dim3(NN / 16, DD / 16), 256, 0, stream>>>(proj, projT16);

    // 5) flash-attention GAT + skip + bias + ELU  (main WMMA kernel)
    gat_attention_kernel<<<dim3(NN / 16, HH), 32, 0, stream>>>(
        projT16, skip, mask, ssrc, stgt, bias, gat_out, gat16);

    // 6) GRU GEMMs (WMMA): gi = gat_out @ w_ih^T + b_ih ; gh = h0 @ w_hh^T + b_hh
    wmma_gemm_f16<<<dim3(G3 / 64, NN / 16), 32, 0, stream>>>(
        gat16, wih16, b_ih, gi, NN, G3, DD);
    wmma_gemm_f16<<<dim3(G3 / 64, NN / 16), 32, 0, stream>>>(
        h016, whh16, b_hh, gh, NN, G3, NGH);

    // 7) gate math -> h_new
    gru_gate_kernel<<<(NN * NGH + 255) / 256, 256, 0, stream>>>(gi, gh, h0, h_new);
}
